// SparseMultiHeadAttention_89567247990905
// MI455X (gfx1250) — matrix operand order repaired and checked by reading, not compiled, not run on hardware
//
#include <hip/hip_runtime.h>

typedef __attribute__((ext_vector_type(16))) _Float16 v16h;
typedef __attribute__((ext_vector_type(8)))  _Float16 v8h;
typedef __attribute__((ext_vector_type(4)))  _Float16 v4h;
typedef __attribute__((ext_vector_type(2)))  _Float16 v2h;
typedef __attribute__((ext_vector_type(8)))  float    v8f;
typedef __attribute__((ext_vector_type(4)))  int      v4i;

#define B_SZ    4
#define S_LEN   1024
#define D_MODEL 1024
#define N_HEADS 16
#define D_K     64

#define GLOBAL_AS __attribute__((address_space(1)))
#define LDS_AS    __attribute__((address_space(3)))

#if defined(__AMDGCN__) && \
    __has_builtin(__builtin_amdgcn_global_load_async_to_lds_b128) && \
    __has_builtin(__builtin_amdgcn_s_wait_asynccnt)
#define HAS_ASYNC 1
#else
#define HAS_ASYNC 0
#endif

// ---------------------------------------------------------------------------
// CDNA5 async global->LDS copy of one 16B chunk (per-lane), ASYNCcnt-tracked.
// Fallback: VGPR-mediated copy (round-1 path).
// ---------------------------------------------------------------------------
static __device__ inline void async_copy16(const _Float16* g, _Float16* l) {
#if HAS_ASYNC
  __builtin_amdgcn_global_load_async_to_lds_b128((GLOBAL_AS v4i*)(g),
                                                 (LDS_AS v4i*)(l), 0, 0);
#else
  *(v8h*)l = *(const v8h*)g;
#endif
}

static __device__ inline void async_wait() {
#if HAS_ASYNC
  __builtin_amdgcn_s_wait_asynccnt(0);
#endif
}

// ---------------------------------------------------------------------------
// WMMA helper: D = A(16x32 f16) * B(32x16 f16) + C(16x16 f32)
// ---------------------------------------------------------------------------
static __device__ inline v8f wmma16(v16h a, v16h b, v8f c) {
  return __builtin_amdgcn_wmma_f32_16x16x32_f16(
      /*neg_a=*/false, a, /*neg_b=*/false, b,
      /*c_mod=*/(short)0, c, /*reuse_a=*/false, /*reuse_b=*/false);
}

// Pack a v16h fragment from two 16-byte LDS chunks (per ISA VGPR layout,
// elements 0..7 -> first chunk, 8..15 -> second chunk).
static __device__ inline v16h pack16(const _Float16* p0, const _Float16* p1) {
  v8h lo = *(const v8h*)p0;
  v8h hs = *(const v8h*)p1;
  v16h r;
#pragma unroll
  for (int i = 0; i < 8; ++i) { r[i] = lo[i]; r[i + 8] = hs[i]; }
  return r;
}

// ---------------------------------------------------------------------------
// fp32 -> f16 conversion (flat layout; used for Q/K/V activations)
// ---------------------------------------------------------------------------
__global__ __launch_bounds__(256) void cvt_f32_f16(const float* __restrict__ src,
                                                   _Float16* __restrict__ dst,
                                                   int n) {
  int i = (blockIdx.x * 256 + threadIdx.x) * 4;
  if (i + 3 < n) {
    float4 v = *(const float4*)(src + i);
    v4h o;
    o[0] = (_Float16)v.x; o[1] = (_Float16)v.y;
    o[2] = (_Float16)v.z; o[3] = (_Float16)v.w;
    *(v4h*)(dst + i) = o;
  }
}

// ---------------------------------------------------------------------------
// fp32 [K][N] -> f16 TRANSPOSED [N][K] (weights).  LDS-tiled 32x32 transpose
// so both global accesses are coalesced.  Pre-transposing the weights once
// makes every GEMM B-tile a straight row-block copy (async-LDS friendly).
// ---------------------------------------------------------------------------
__global__ __launch_bounds__(256) void cvt_w_f16t(const float* __restrict__ W,
                                                  _Float16* __restrict__ Wt) {
  __shared__ float tile[32][33];
  const int n0 = blockIdx.x * 32;
  const int k0 = blockIdx.y * 32;
  const int tx = threadIdx.x & 31;
  const int ty = threadIdx.x >> 5;  // 0..7
#pragma unroll
  for (int i = 0; i < 4; ++i) {
    int k = ty + i * 8;
    tile[k][tx] = W[(size_t)(k0 + k) * D_MODEL + n0 + tx];
  }
  __syncthreads();
#pragma unroll
  for (int i = 0; i < 4; ++i) {
    int n = ty + i * 8;
    Wt[(size_t)(n0 + n) * D_MODEL + k0 + tx] = (_Float16)tile[tx][n];
  }
}

// ---------------------------------------------------------------------------
// Stage one 128x32 f16 tile (rows r0..r0+127, cols k0..k0+31 of a [X][1024]
// row-major matrix) into LDS with 40-half row stride.  512 16B chunks,
// 2 per thread (256 threads), issued as async global->LDS copies.
// ---------------------------------------------------------------------------
static __device__ inline void stage_tile_256(const _Float16* __restrict__ src,
                                             _Float16* dst, int r0, int k0) {
  const int tid = threadIdx.x;
#pragma unroll
  for (int i = 0; i < 2; ++i) {
    int e = tid + i * 256;   // 0..511
    int m = e >> 2;          // 0..127
    int c = (e & 3) << 3;    // 0,8,16,24 halfs
    async_copy16(&src[(size_t)(r0 + m) * D_MODEL + k0 + c], &dst[m * 40 + c]);
  }
}

// ---------------------------------------------------------------------------
// 128x128 GEMM tile engine, double-buffered LDS, async staging.
// A:[4096][1024] f16 row-major, Wt:[1024][1024] f16 (pre-transposed [N][K]).
// Block = 256 threads (8 waves); wave w owns rows w*16..+15, all 128 cols.
// ---------------------------------------------------------------------------
static __device__ inline void gemm_tile_128x128(const _Float16* __restrict__ A,
                                                const _Float16* __restrict__ Wt,
                                                int m0, int n0,
                                                _Float16* As0, _Float16* As1,
                                                _Float16* Bt0, _Float16* Bt1,
                                                v8f acc[8]) {
  const int tid  = threadIdx.x;
  const int wave = tid >> 5;
  const int lane = tid & 31;
  const int ln   = lane & 15;
  const int hi   = lane >> 4;

  { v8f z = {};
#pragma unroll
    for (int f = 0; f < 8; ++f) acc[f] = z; }

  stage_tile_256(A,  As0, m0, 0);
  stage_tile_256(Wt, Bt0, n0, 0);
  async_wait();
  __syncthreads();

  for (int k0 = 0; k0 < 1024; k0 += 32) {
    const int cur = (k0 >> 5) & 1;
    _Float16* Asc = cur ? As1 : As0;
    _Float16* Btc = cur ? Bt1 : Bt0;
    if (k0 + 32 < 1024) {  // prefetch next tiles into the other buffer
      stage_tile_256(A,  cur ? As0 : As1, m0, k0 + 32);
      stage_tile_256(Wt, cur ? Bt0 : Bt1, n0, k0 + 32);
    }
    // A fragment: row = wave*16 + ln, K chunks at hi*8 and hi*8+16
    const _Float16* ap = &Asc[(wave * 16 + ln) * 40 + hi * 8];
    v16h afrag = pack16(ap, ap + 16);
    // Preload all 8 B fragments, then back-to-back WMMAs
    v16h bfr[8];
#pragma unroll
    for (int f = 0; f < 8; ++f) {
      const _Float16* bp = &Btc[(f * 16 + ln) * 40 + hi * 8];
      bfr[f] = pack16(bp, bp + 16);
    }
#pragma unroll
    for (int f = 0; f < 8; ++f) acc[f] = wmma16(afrag, bfr[f], acc[f]);
    async_wait();
    __syncthreads();
  }
}

// ---------------------------------------------------------------------------
// QKV projection: grid.z selects q/k/v.  q,k -> head-major [B][H][S][D_K];
// v -> TRANSPOSED head-major [B][H][D_K][S] so the attention V-tile is a
// straight (async-copyable) row block.
// ---------------------------------------------------------------------------
__global__ __launch_bounds__(256) void gemm_qkv_kernel(
    const _Float16* __restrict__ Xq, const _Float16* __restrict__ Xk,
    const _Float16* __restrict__ Xv, const _Float16* __restrict__ Wqt,
    const _Float16* __restrict__ Wkt, const _Float16* __restrict__ Wvt,
    const float* __restrict__ bq, const float* __restrict__ bk,
    const float* __restrict__ bv, _Float16* __restrict__ qd,
    _Float16* __restrict__ kd, _Float16* __restrict__ vd) {
  __shared__ _Float16 As0[128 * 40], As1[128 * 40];
  __shared__ _Float16 Bt0[128 * 40], Bt1[128 * 40];
  const int z = blockIdx.z;
  const _Float16* X  = (z == 0) ? Xq  : (z == 1) ? Xk  : Xv;
  const _Float16* Wt = (z == 0) ? Wqt : (z == 1) ? Wkt : Wvt;
  const float* bias  = (z == 0) ? bq  : (z == 1) ? bk  : bv;
  _Float16* dst      = (z == 0) ? qd  : (z == 1) ? kd  : vd;

  const int n0 = blockIdx.x * 128;
  const int m0 = blockIdx.y * 128;
  v8f acc[8];
  gemm_tile_128x128(X, Wt, m0, n0, As0, As1, Bt0, Bt1, acc);

  const int tid = threadIdx.x, wave = tid >> 5, lane = tid & 31;
  const int ln = lane & 15, hi = lane >> 4;
#pragma unroll
  for (int f = 0; f < 8; ++f) {
    const int col = n0 + f * 16 + ln;
    const int h = col >> 6, d = col & 63;
    const float bb = bias[col];
#pragma unroll
    for (int r = 0; r < 8; ++r) {
      const int row  = m0 + wave * 16 + hi * 8 + r;
      const int bidx = row >> 10, s = row & 1023;
      const _Float16 val = (_Float16)(acc[f][r] + bb);
      if (z == 2)  // V transposed: [B][H][D_K][S]
        dst[(((size_t)bidx * N_HEADS + h) * D_K + d) * S_LEN + s] = val;
      else         // Q,K: [B][H][S][D_K]
        dst[(((size_t)bidx * N_HEADS + h) * S_LEN + s) * D_K + d] = val;
    }
  }
}

// ---------------------------------------------------------------------------
// Output projection: attn_ws @ Wo + bo -> fp32 d_out
// ---------------------------------------------------------------------------
__global__ __launch_bounds__(256) void gemm_out_kernel(
    const _Float16* __restrict__ A, const _Float16* __restrict__ Wt,
    const float* __restrict__ bias, float* __restrict__ out) {
  __shared__ _Float16 As0[128 * 40], As1[128 * 40];
  __shared__ _Float16 Bt0[128 * 40], Bt1[128 * 40];
  const int n0 = blockIdx.x * 128;
  const int m0 = blockIdx.y * 128;
  v8f acc[8];
  gemm_tile_128x128(A, Wt, m0, n0, As0, As1, Bt0, Bt1, acc);

  const int tid = threadIdx.x, wave = tid >> 5, lane = tid & 31;
  const int ln = lane & 15, hi = lane >> 4;
#pragma unroll
  for (int f = 0; f < 8; ++f) {
    const int col = n0 + f * 16 + ln;
    const float bb = bias[col];
#pragma unroll
    for (int r = 0; r < 8; ++r) {
      const int row = m0 + wave * 16 + hi * 8 + r;
      out[(size_t)row * D_MODEL + col] = acc[f][r] + bb;
    }
  }
}

// ---------------------------------------------------------------------------
// Flash attention with strided mask.  One block per (b, h, 64-query tile).
// 4 waves x 16 query rows.  Q/K tiles row-major [i][d]; V tile comes from
// the pre-transposed layout so [d][j] is a straight async copy.
// ---------------------------------------------------------------------------
__global__ __launch_bounds__(128) void attn_kernel(
    const _Float16* __restrict__ qg, const _Float16* __restrict__ kg,
    const _Float16* __restrict__ vgT, _Float16* __restrict__ og) {
  const int qb = blockIdx.x;
  const int h  = blockIdx.y;
  const int b  = blockIdx.z;
  const int tid  = threadIdx.x;
  const int wave = tid >> 5;
  const int lane = tid & 31;
  const int ln   = lane & 15;
  const int hi   = lane >> 4;

  __shared__ _Float16 qs[64 * 72];     // Q tile  [i][d]
  __shared__ _Float16 ks[64 * 72];     // K tile  [j][d]
  __shared__ _Float16 vt[64 * 72];     // V tile  [d][j]  (from vT layout)
  __shared__ _Float16 ps[4][16 * 72];  // per-wave P (C->A relayout)

  const size_t hb = ((size_t)b * N_HEADS + h) * S_LEN * D_K;
  const _Float16* qh = qg + hb;
  const _Float16* kh = kg + hb;
  const _Float16* vh = vgT + hb;       // [D_K][S_LEN] per head
  const int q0 = qb * 64;

  // stage Q tile: 64 rows x 64 halfs = 512 16B chunks, 4 per thread
#pragma unroll
  for (int i = 0; i < 4; ++i) {
    int e = tid + i * 128;       // 0..511
    int r = e >> 3, c = (e & 7) << 3;
    async_copy16(&qh[(size_t)(q0 + r) * D_K + c], &qs[r * 72 + c]);
  }
  async_wait();
  __syncthreads();

  v16h qa[2];
#pragma unroll
  for (int kstep = 0; kstep < 2; ++kstep) {
    const _Float16* p = &qs[(wave * 16 + ln) * 72 + kstep * 32 + hi * 8];
    qa[kstep] = pack16(p, p + 16);
  }

  float mrow[8], lrow[8];
#pragma unroll
  for (int r = 0; r < 8; ++r) { mrow[r] = -1e30f; lrow[r] = 0.0f; }
  v8f oacc[4];
  { v8f z = {};
#pragma unroll
    for (int f = 0; f < 4; ++f) oacc[f] = z; }

  _Float16* psw = &ps[wave][0];

  for (int kb = 0; kb <= qb; ++kb) {
    __syncthreads();
#pragma unroll
    for (int i = 0; i < 4; ++i) {
      int e = tid + i * 128;
      int r = e >> 3, c = (e & 7) << 3;
      async_copy16(&kh[(size_t)(kb * 64 + r) * D_K + c], &ks[r * 72 + c]);
      async_copy16(&vh[(size_t)r * S_LEN + kb * 64 + c], &vt[r * 72 + c]);
    }
    async_wait();
    __syncthreads();

    // scores = q @ k^T : 8 WMMAs -> 16x64 fp32
    v8f sf[4];
    { v8f z = {};
#pragma unroll
      for (int f = 0; f < 4; ++f) sf[f] = z; }
#pragma unroll
    for (int f = 0; f < 4; ++f)
#pragma unroll
      for (int kstep = 0; kstep < 2; ++kstep) {
        const _Float16* p = &ks[(f * 16 + ln) * 72 + kstep * 32 + hi * 8];
        sf[f] = wmma16(qa[kstep], pack16(p, p + 16), sf[f]);
      }

    // strided mask + online softmax
    const int ibase = q0 + wave * 16 + hi * 8;
    const int jbase = kb * 64 + ln;
    float alpha[8];
#pragma unroll
    for (int r = 0; r < 8; ++r) {
      float mx = -1e30f;
#pragma unroll
      for (int f = 0; f < 4; ++f) {
        const int dd = (ibase + r) - (jbase + f * 16);
        float sv = sf[f][r] * 0.125f;  // 1/sqrt(64)
        const bool ok = (dd >= 0) && ((dd <= 3) || ((dd & 3) == 0));
        sv = ok ? sv : -1e30f;
        sf[f][r] = sv;
        mx = fmaxf(mx, sv);
      }
#pragma unroll
      for (int m = 1; m < 16; m <<= 1) mx = fmaxf(mx, __shfl_xor(mx, m, 32));
      const float mnew = fmaxf(mrow[r], mx);
      alpha[r] = __expf(mrow[r] - mnew);
      mrow[r]  = mnew;

      float rs = 0.0f;
#pragma unroll
      for (int f = 0; f < 4; ++f) {
        const float pv = __expf(sf[f][r] - mnew);
        rs += pv;
        psw[(r + hi * 8) * 72 + f * 16 + ln] = (_Float16)pv;
      }
#pragma unroll
      for (int m = 1; m < 16; m <<= 1) rs += __shfl_xor(rs, m, 32);
      lrow[r] = lrow[r] * alpha[r] + rs;
    }

#pragma unroll
    for (int f = 0; f < 4; ++f)
#pragma unroll
      for (int r = 0; r < 8; ++r) oacc[f][r] *= alpha[r];

    // O += P @ V
#pragma unroll
    for (int kstep = 0; kstep < 2; ++kstep) {
      const _Float16* pa = &psw[ln * 72 + kstep * 32 + hi * 8];
      v16h af = pack16(pa, pa + 16);
#pragma unroll
      for (int f = 0; f < 4; ++f) {
        const _Float16* pb = &vt[(f * 16 + ln) * 72 + kstep * 32 + hi * 8];
        oacc[f] = wmma16(af, pack16(pb, pb + 16), oacc[f]);
      }
    }
  }

  // epilogue: normalize, write concat-head layout [B][S][D]
#pragma unroll
  for (int f = 0; f < 4; ++f) {
#pragma unroll
    for (int r = 0; r < 8; ++r) {
      const int i   = q0 + wave * 16 + hi * 8 + r;
      const int col = h * D_K + f * 16 + ln;
      og[((size_t)b * S_LEN + i) * D_MODEL + col] =
          (_Float16)(oacc[f][r] / lrow[r]);
    }
  }
}

// ---------------------------------------------------------------------------
// Host-side launch
// ---------------------------------------------------------------------------
extern "C" void kernel_launch(void* const* d_in, const int* in_sizes, int n_in,
                              void* d_out, int out_size, void* d_ws,
                              size_t ws_size, hipStream_t stream) {
  const float* Q  = (const float*)d_in[0];
  const float* K  = (const float*)d_in[1];
  const float* V  = (const float*)d_in[2];
  const float* Wq = (const float*)d_in[3];
  const float* bq = (const float*)d_in[4];
  const float* Wk = (const float*)d_in[5];
  const float* bk = (const float*)d_in[6];
  const float* Wv = (const float*)d_in[7];
  const float* bv = (const float*)d_in[8];
  const float* Wo = (const float*)d_in[9];
  const float* bo = (const float*)d_in[10];
  float* out = (float*)d_out;

  _Float16* ws16 = (_Float16*)d_ws;
  const size_t MD = (size_t)B_SZ * S_LEN * D_MODEL;  // 4M elems
  const size_t WW = (size_t)D_MODEL * D_MODEL;       // 1M elems
  _Float16* Xq  = ws16;
  _Float16* Xk  = Xq + MD;
  _Float16* Xv  = Xk + MD;
  _Float16* Wqt = Xv + MD;      // pre-transposed f16 weights [N][K]
  _Float16* Wkt = Wqt + WW;
  _Float16* Wvt = Wkt + WW;
  _Float16* Wot = Wvt + WW;
  _Float16* qws = Wot + WW;     // [B][H][S][D_K]
  _Float16* kws = qws + MD;
  _Float16* vwsT = kws + MD;    // [B][H][D_K][S]
  _Float16* aws = vwsT + MD;    // [B][S][D]

  const int gMD = (int)(MD / 4 / 256);  // 4096 blocks
  cvt_f32_f16<<<gMD, 256, 0, stream>>>(Q, Xq, (int)MD);
  cvt_f32_f16<<<gMD, 256, 0, stream>>>(K, Xk, (int)MD);
  cvt_f32_f16<<<gMD, 256, 0, stream>>>(V, Xv, (int)MD);

  dim3 gT(D_MODEL / 32, D_MODEL / 32);
  cvt_w_f16t<<<gT, 256, 0, stream>>>(Wq, Wqt);
  cvt_w_f16t<<<gT, 256, 0, stream>>>(Wk, Wkt);
  cvt_w_f16t<<<gT, 256, 0, stream>>>(Wv, Wvt);
  cvt_w_f16t<<<gT, 256, 0, stream>>>(Wo, Wot);

  gemm_qkv_kernel<<<dim3(D_MODEL / 128, (B_SZ * S_LEN) / 128, 3), 256, 0,
                    stream>>>(Xq, Xk, Xv, Wqt, Wkt, Wvt, bq, bk, bv, qws, kws,
                              vwsT);

  attn_kernel<<<dim3(S_LEN / 64, N_HEADS, B_SZ), 128, 0, stream>>>(qws, kws,
                                                                   vwsT, aws);

  gemm_out_kernel<<<dim3(D_MODEL / 128, (B_SZ * S_LEN) / 128, 1), 256, 0,
                    stream>>>(aws, Wot, bo, out);
}
